// PatchCore_764504179304
// MI455X (gfx1250) — matrix operand order repaired and checked by reading, not compiled, not run on hardware
//
#include <hip/hip_runtime.h>
#include <stdint.h>

typedef __attribute__((ext_vector_type(2)))  float  v2f;
typedef __attribute__((ext_vector_type(8)))  float  v8f;
typedef __attribute__((ext_vector_type(8)))  __bf16 v8bf;
typedef __attribute__((ext_vector_type(16))) __bf16 v16bf;

#define C_DIM 128
#define FTILE 16      // f rows per wave
#define WAVES 8       // waves per block
#define MBLK 16       // m columns per WMMA tile
#define MCHUNK 512    // m columns per workgroup (grid.y)

// ---------------------------------------------------------------------------
// bf16 helpers (round-to-nearest-even), bit-level so no header dependency.
// ---------------------------------------------------------------------------
__device__ __forceinline__ unsigned short f2bf_rne(float x) {
  unsigned int u = __float_as_uint(x);
  u += 0x7FFFu + ((u >> 16) & 1u);
  return (unsigned short)(u >> 16);
}
__device__ __forceinline__ float bf2f(unsigned short b) {
  return __uint_as_float(((unsigned int)b) << 16);
}

// ---------------------------------------------------------------------------
// Pre-pass: squared L2 norm of each row (one wave32 per row), from exact f32.
// ---------------------------------------------------------------------------
__global__ __launch_bounds__(256) void row_norms_kernel(
    const float* __restrict__ f, const float* __restrict__ m,
    float* __restrict__ f2, float* __restrict__ m2, int N, int M) {
  int gwave = (int)((blockIdx.x * blockDim.x + threadIdx.x) >> 5);
  int lane  = (int)(threadIdx.x & 31);
  if (gwave >= N + M) return;
  const float* src = (gwave < N) ? (f + (size_t)gwave * C_DIM)
                                 : (m + (size_t)(gwave - N) * C_DIM);
  float4 v = *(const float4*)(src + lane * 4);
  float s = v.x * v.x + v.y * v.y + v.z * v.z + v.w * v.w;
  s += __shfl_xor(s, 1);
  s += __shfl_xor(s, 2);
  s += __shfl_xor(s, 4);
  s += __shfl_xor(s, 8);
  s += __shfl_xor(s, 16);
  if (lane == 0) {
    if (gwave < N) f2[gwave] = s;
    else           m2[gwave - N] = s;
  }
}

// ---------------------------------------------------------------------------
// Pre-pass: init output to +inf bits for atomic-min combining.
// ---------------------------------------------------------------------------
__global__ __launch_bounds__(256) void init_out_kernel(uint32_t* __restrict__ out, int n) {
  int i = (int)(blockIdx.x * blockDim.x + threadIdx.x);
  if (i < n) out[i] = 0x7F800000u;
}

// ---------------------------------------------------------------------------
// Pre-pass: split f32 -> bf16 hi/lo pair (x ~= hi + lo, ~16-17 mantissa bits).
// One thread per float4; writes packed ushort4 (8B) to hi and lo arrays.
// ---------------------------------------------------------------------------
__global__ __launch_bounds__(256) void split_bf16_kernel(
    const float* __restrict__ f, const float* __restrict__ m,
    unsigned short* __restrict__ fh, unsigned short* __restrict__ fl,
    unsigned short* __restrict__ mh, unsigned short* __restrict__ ml,
    int nf4, int nm4) {
  int i = (int)(blockIdx.x * blockDim.x + threadIdx.x);
  const float4* src;
  unsigned short *dh, *dl;
  int idx;
  if (i < nf4)            { src = (const float4*)f; dh = fh; dl = fl; idx = i; }
  else if (i < nf4 + nm4) { src = (const float4*)m; dh = mh; dl = ml; idx = i - nf4; }
  else return;
  float4 v = src[idx];
  unsigned short h0 = f2bf_rne(v.x), h1 = f2bf_rne(v.y),
                 h2 = f2bf_rne(v.z), h3 = f2bf_rne(v.w);
  unsigned short l0 = f2bf_rne(v.x - bf2f(h0)), l1 = f2bf_rne(v.y - bf2f(h1)),
                 l2 = f2bf_rne(v.z - bf2f(h2)), l3 = f2bf_rne(v.w - bf2f(h3));
  uint2 hp, lp;
  hp.x = (uint32_t)h0 | ((uint32_t)h1 << 16);
  hp.y = (uint32_t)h2 | ((uint32_t)h3 << 16);
  lp.x = (uint32_t)l0 | ((uint32_t)l1 << 16);
  lp.y = (uint32_t)l2 | ((uint32_t)l3 << 16);
  ((uint2*)dh)[idx] = hp;
  ((uint2*)dl)[idx] = lp;
}

// ---------------------------------------------------------------------------
// Main (bf16-split path): per wave, 16 f-rows resident as A hi/lo fragments;
// sweep 512 m columns in 16-col blocks, x2 unrolled so the scheduler has
// 24 independent WMMAs + 32 B-loads per iteration to hide the WMMA->VALU
// fold hazard and keep the loadcnt pipeline deep.
//
// 16-bit WMMA k order used here (wave32; contiguous, same for A and B):
//   A 16x32: lane l = row (l&15); elements 0-15 = K 16*(l>>4)..+15
//            -> two aligned b128 loads.
//   B 32x16: lane l = col (l&15); elements 0-15 = K 16*(l>>4)..+15
//            -> one contiguous aligned 32B run.
// ---------------------------------------------------------------------------
__global__ __launch_bounds__(256) void patchcore_bf16_kernel(
    const unsigned short* __restrict__ fh, const unsigned short* __restrict__ fl,
    const unsigned short* __restrict__ mh, const unsigned short* __restrict__ ml,
    const float* __restrict__ f2, const float* __restrict__ m2,
    float* __restrict__ out) {
  const int lane  = (int)(threadIdx.x & 31);
  const int wave  = (int)(threadIdx.x >> 5);
  const int row16 = lane & 15;
  const int hi    = lane >> 4;
  const int fbase = (int)blockIdx.x * (FTILE * WAVES) + wave * FTILE;

  const unsigned short* fhrow = fh + (size_t)(fbase + row16) * C_DIM;
  const unsigned short* flrow = fl + (size_t)(fbase + row16) * C_DIM;

  v16bf ah[4], al[4];
#pragma unroll
  for (int k = 0; k < 4; ++k) {
    const int c0 = k * 32 + 16 * hi;
    const int c1 = c0 + 8;
    v8bf h0 = *(const v8bf*)(fhrow + c0);
    v8bf h1 = *(const v8bf*)(fhrow + c1);
    ah[k] = __builtin_shufflevector(h0, h1, 0, 1, 2, 3, 4, 5, 6, 7,
                                            8, 9, 10, 11, 12, 13, 14, 15);
    v8bf q0 = *(const v8bf*)(flrow + c0);
    v8bf q1 = *(const v8bf*)(flrow + c1);
    al[k] = __builtin_shufflevector(q0, q1, 0, 1, 2, 3, 4, 5, 6, 7,
                                            8, 9, 10, 11, 12, 13, 14, 15);
  }

  float minacc[8];
#pragma unroll
  for (int v = 0; v < 8; ++v) minacc[v] = __builtin_inff();

  const int mchunk = (int)blockIdx.y * MCHUNK;
#pragma unroll 2
  for (int mb = mchunk; mb < mchunk + MCHUNK; mb += MBLK) {
    const unsigned short* mhrow = mh + (size_t)(mb + row16) * C_DIM + 16 * hi;
    const unsigned short* mlrow = ml + (size_t)(mb + row16) * C_DIM + 16 * hi;
    if (mb + 2 * MBLK < mchunk + MCHUNK) {     // wave-uniform, 2 blocks ahead
      __builtin_prefetch(mhrow + 2 * MBLK * C_DIM, 0, 1);
      __builtin_prefetch(mlrow + 2 * MBLK * C_DIM, 0, 1);
    }
    v16bf bh[4], bl[4];
#pragma unroll
    for (int k = 0; k < 4; ++k) {
      bh[k] = *(const v16bf*)(mhrow + k * 32);
      bl[k] = *(const v16bf*)(mlrow + k * 32);
    }
    v8f acc0 = {}, acc1 = {}, acc2 = {};
#pragma unroll
    for (int k = 0; k < 4; ++k) {
      acc0 = __builtin_amdgcn_wmma_f32_16x16x32_bf16(
          false, ah[k], false, bh[k], (short)0, acc0, false, false);
      acc1 = __builtin_amdgcn_wmma_f32_16x16x32_bf16(
          false, ah[k], false, bl[k], (short)0, acc1, false, false);
      acc2 = __builtin_amdgcn_wmma_f32_16x16x32_bf16(
          false, al[k], false, bh[k], (short)0, acc2, false, false);
    }
    const float m2s = m2[mb + row16];
#pragma unroll
    for (int v = 0; v < 8; ++v) {
      float cr = acc0[v] + (acc1[v] + acc2[v]);
      minacc[v] = fminf(minacc[v], fmaf(-2.0f, cr, m2s));
    }
  }

  // C/D layout: lane l, VGPR v -> (row = v + 8*(l>>4), col = l&15).
#pragma unroll
  for (int v = 0; v < 8; ++v) {
    float r = minacc[v];
    r = fminf(r, __shfl_xor(r, 1));
    r = fminf(r, __shfl_xor(r, 2));
    r = fminf(r, __shfl_xor(r, 4));
    r = fminf(r, __shfl_xor(r, 8));
    if (row16 == 0) {
      int row = fbase + v + 8 * hi;
      float dist = sqrtf(fmaxf(f2[row] + r, 0.0f));
      atomicMin((uint32_t*)out + row, __float_as_uint(dist));
    }
  }
}

// ---------------------------------------------------------------------------
// Fallback (exact-f32 path, used only if ws can't hold the bf16 copies):
// identical structure built on v_wmma_f32_16x16x4_f32.
// ---------------------------------------------------------------------------
__global__ __launch_bounds__(256, 2) void patchcore_f32_kernel(
    const float* __restrict__ f, const float* __restrict__ m,
    const float* __restrict__ f2, const float* __restrict__ m2,
    float* __restrict__ out) {
  const int lane  = (int)(threadIdx.x & 31);
  const int wave  = (int)(threadIdx.x >> 5);
  const int row16 = lane & 15;
  const int hi    = lane >> 4;
  const int fbase = (int)blockIdx.x * (FTILE * WAVES) + wave * FTILE;

  const float* frow = f + (size_t)(fbase + row16) * C_DIM + 2 * hi;
  v2f a[32];
#pragma unroll
  for (int s = 0; s < 32; ++s) a[s] = *(const v2f*)(frow + s * 4);

  float minacc[8];
#pragma unroll
  for (int v = 0; v < 8; ++v) minacc[v] = __builtin_inff();

  const int mchunk = (int)blockIdx.y * MCHUNK;
  for (int mb = mchunk; mb < mchunk + MCHUNK; mb += MBLK) {
    const float* mrow = m + (size_t)(mb + row16) * C_DIM + 2 * hi;
    if (mb + MBLK < mchunk + MCHUNK)
      __builtin_prefetch(mrow + MBLK * C_DIM, 0, 1);
    v2f b[32];
#pragma unroll
    for (int s = 0; s < 32; ++s) b[s] = *(const v2f*)(mrow + s * 4);
    v8f acc0 = {}, acc1 = {};
#pragma unroll
    for (int s = 0; s < 32; s += 2) {
      acc0 = __builtin_amdgcn_wmma_f32_16x16x4_f32(
          false, a[s], false, b[s], (short)0, acc0, false, false);
      acc1 = __builtin_amdgcn_wmma_f32_16x16x4_f32(
          false, a[s + 1], false, b[s + 1], (short)0, acc1, false, false);
    }
    const float m2s = m2[mb + row16];
#pragma unroll
    for (int v = 0; v < 8; ++v)
      minacc[v] = fminf(minacc[v], fmaf(-2.0f, acc0[v] + acc1[v], m2s));
  }
#pragma unroll
  for (int v = 0; v < 8; ++v) {
    float r = minacc[v];
    r = fminf(r, __shfl_xor(r, 1));
    r = fminf(r, __shfl_xor(r, 2));
    r = fminf(r, __shfl_xor(r, 4));
    r = fminf(r, __shfl_xor(r, 8));
    if (row16 == 0) {
      int row = fbase + v + 8 * hi;
      float dist = sqrtf(fmaxf(f2[row] + r, 0.0f));
      atomicMin((uint32_t*)out + row, __float_as_uint(dist));
    }
  }
}

// ---------------------------------------------------------------------------
extern "C" void kernel_launch(void* const* d_in, const int* in_sizes, int n_in,
                              void* d_out, int out_size, void* d_ws, size_t ws_size,
                              hipStream_t stream) {
  const float* f = (const float*)d_in[0];   // [N,1,1,C] -> N x 128
  const float* m = (const float*)d_in[1];   // [1,M,1,C] -> M x 128
  float* out = (float*)d_out;               // [N,1] -> N floats

  const int N = in_sizes[0] / C_DIM;        // 8192
  const int M = in_sizes[1] / C_DIM;        // 16384

  // Workspace layout: f2[N] f32 | m2[M] f32 | fh | fl | mh | ml (bf16)
  float* f2 = (float*)d_ws;
  float* m2 = f2 + N;
  unsigned short* fh = (unsigned short*)(m2 + M);
  unsigned short* fl = fh + (size_t)N * C_DIM;
  unsigned short* mh = fl + (size_t)N * C_DIM;
  unsigned short* ml = mh + (size_t)M * C_DIM;
  const size_t need_bf16 = (size_t)(N + M) * 4u + (size_t)(N + M) * C_DIM * 4u;

  // 1) Row norms (exact f32): one wave per row.
  {
    int rows = N + M;
    row_norms_kernel<<<(rows + WAVES - 1) / WAVES, 256, 0, stream>>>(f, m, f2, m2, N, M);
  }
  // 2) Init output to +inf.
  init_out_kernel<<<(N + 255) / 256, 256, 0, stream>>>((uint32_t*)out, N);

  dim3 grid(N / (FTILE * WAVES), M / MCHUNK);
  if (ws_size >= need_bf16) {
    // 3a) Split into bf16 hi/lo copies, then bf16-split WMMA main kernel.
    int nf4 = N * (C_DIM / 4), nm4 = M * (C_DIM / 4);
    int tot = nf4 + nm4;
    split_bf16_kernel<<<(tot + 255) / 256, 256, 0, stream>>>(f, m, fh, fl, mh, ml, nf4, nm4);
    patchcore_bf16_kernel<<<grid, 256, 0, stream>>>(fh, fl, mh, ml, f2, m2, out);
  } else {
    // 3b) Exact-f32 WMMA fallback (no extra workspace).
    patchcore_f32_kernel<<<grid, 256, 0, stream>>>(f, m, f2, m2, out);
  }
}
